// PredatorCritic_82154134438569
// MI455X (gfx1250) — matrix operand order repaired and checked by reading, not compiled, not run on hardware
//
#include <hip/hip_runtime.h>
#include <hip/hip_bf16.h>

// Problem constants (match reference)
static constexpr int Nn = 30000;
static constexpr int Ee = 600000;
static constexpr int Pp = 3000;

typedef __attribute__((ext_vector_type(16))) _Float16 v16h;
typedef __attribute__((ext_vector_type(8)))  float    v8f;

__device__ __forceinline__ float eluf(float v) {
    return v > 0.0f ? v : (__expf(v) - 1.0f);
}
__device__ __forceinline__ float leaky01(float v) {
    return v >= 0.0f ? v : 0.1f * v;
}
// order-preserving float<->uint key for atomicMax over signed floats
__device__ __forceinline__ unsigned fkey(float f) {
    unsigned u = __float_as_uint(f);
    return (u & 0x80000000u) ? ~u : (u | 0x80000000u);
}
__device__ __forceinline__ float funkey(unsigned k) {
    unsigned u = (k & 0x80000000u) ? (k & 0x7FFFFFFFu) : ~k;
    return __uint_as_float(u);
}

// ---------------------------------------------------------------- zero
__global__ void zero_kernel(float* __restrict__ p, int n) {
    int i = blockIdx.x * blockDim.x + threadIdx.x;
    if (i < n) p[i] = 0.0f;
}

// ---------------------------------------------------------------- action MLPs: a1 (P,16), a2 (P,64)
__global__ void action_kernel(const float* __restrict__ action,
                              const float* __restrict__ a1_w1, const float* __restrict__ a1_b1,
                              const float* __restrict__ a1_w2, const float* __restrict__ a1_b2,
                              const float* __restrict__ a2_w1, const float* __restrict__ a2_b1,
                              const float* __restrict__ a2_w2, const float* __restrict__ a2_b2,
                              float* __restrict__ a1o, float* __restrict__ a2o) {
    int i = blockIdx.x * blockDim.x + threadIdx.x;
    if (i >= Pp) return;
    float ang = action[i] * 3.14159265358979323846f;
    float sa = __sinf(ang), ca = __cosf(ang);

    float h1v[16];
#pragma unroll
    for (int j = 0; j < 16; ++j)
        h1v[j] = eluf(fmaf(sa, a1_w1[j], fmaf(ca, a1_w1[16 + j], a1_b1[j])));
    for (int o = 0; o < 16; ++o) {
        float s = a1_b2[o];
#pragma unroll
        for (int j = 0; j < 16; ++j) s = fmaf(h1v[j], a1_w2[j * 16 + o], s);
        a1o[i * 16 + o] = s;
    }
    float h2v[64];
#pragma unroll
    for (int j = 0; j < 64; ++j)
        h2v[j] = eluf(fmaf(sa, a2_w1[j], fmaf(ca, a2_w1[64 + j], a2_b1[j])));
    for (int o = 0; o < 64; ++o) {
        float s = a2_b2[o];
#pragma unroll
        for (int j = 0; j < 64; ++j) s = fmaf(h2v[j], a2_w2[j * 64 + o], s);
        a2o[i * 64 + o] = s;
    }
}

// ---------------------------------------------------------------- node embedding h0 (N,16) incl. +a1 on [:P]
__global__ void node_embed_kernel(const float* __restrict__ x, const int* __restrict__ mask,
                                  const float* __restrict__ pe_w1, const float* __restrict__ pe_b1,
                                  const float* __restrict__ pe_w2, const float* __restrict__ pe_b2,
                                  const float* __restrict__ ent_emb, const float* __restrict__ a1,
                                  float* __restrict__ h0) {
    int n = blockIdx.x * blockDim.x + threadIdx.x;
    if (n >= Nn) return;
    float x0 = x[n * 3 + 0], x1 = x[n * 3 + 1], x2 = x[n * 3 + 2];
    float hid[16];
#pragma unroll
    for (int j = 0; j < 16; ++j)
        hid[j] = eluf(fmaf(x0, pe_w1[j], fmaf(x1, pe_w1[16 + j], fmaf(x2, pe_w1[32 + j], pe_b1[j]))));
    int mk = mask[n];
    for (int o = 0; o < 16; ++o) {
        float s = pe_b2[o] + ent_emb[mk * 16 + o];
#pragma unroll
        for (int j = 0; j < 16; ++j) s = fmaf(hid[j], pe_w2[j * 16 + o], s);
        if (n < Pp) s += a1[n * 16 + o];
        h0[n * 16 + o] = s;
    }
}

// ---------------------------------------------------------------- degree
__global__ void deg_kernel(const int* __restrict__ colp, float* __restrict__ deg) {
    int e = blockIdx.x * blockDim.x + threadIdx.x;
    if (e < Ee) atomicAdd(&deg[colp[e]], 1.0f);
}

// ---------------------------------------------------------------- WMMA edge-message kernel
// msg[e,o] = sum_{j,i} t[e,j]*h0[row[e],i]*nn_w2[j, i*16+o] + sum_i h0_i*nn_b2[i*16+o]
// 16-edge tile: A = U (16 x 288 fp16, built in f16 math), B = packed W2p (288 x 16 fp16, constant),
// 9x v_wmma_f32_16x16x32_f16 per tile, f32 atomic scatter into agg[col].
__launch_bounds__(128)
__global__ void edge_msg_kernel(const float* __restrict__ x,
                                const int* __restrict__ rowp, const int* __restrict__ colp,
                                const int* __restrict__ edge_attr,
                                const float* __restrict__ et_emb,
                                const float* __restrict__ nn_w1, const float* __restrict__ nn_b1,
                                const float* __restrict__ nn_w2, const float* __restrict__ nn_b2,
                                const float* __restrict__ h0,
                                float* __restrict__ agg) {
    __shared__ float s_w1[12 * 16];
    __shared__ float s_b1[16];
    __shared__ float s_t[4][16][16];   // per-wave t tiles
    __shared__ int   s_c[4][16];       // per-wave col indices

    const int tid = threadIdx.x;
    for (int i = tid; i < 192; i += blockDim.x) s_w1[i] = nn_w1[i];
    if (tid < 16) s_b1[tid] = nn_b1[tid];
    __syncthreads();

    const int lane = tid & 31;
    const int w    = tid >> 5;
    const int half = lane >> 4;        // 0: K-low half, 1: K-high half
    const int m    = lane & 15;        // edge row within tile (A) / output col (B,C)
    const int o    = lane & 15;

    // Preload constant B fragments: W2p[k,o], k = j*16+i (k<256), bias rows k=256..271, zero pad to 288.
    // B layout (16x16x32 f16): lanes 0-15 -> N=o, K=kb*32+0..7,16..23 ; lanes 16-31 -> N=o, K=kb*32+8..15,24..31.
    v16h bfrag[9];
    {
        const int kbase0 = half ? 8 : 0;
#pragma unroll
        for (int kb = 0; kb < 9; ++kb) {
#pragma unroll
            for (int q = 0; q < 16; ++q) {
                int k = kb * 32 + kbase0 + (q & 7) + (q >> 3) * 16;
                float v;
                if (k < 256)      v = nn_w2[(k >> 4) * 256 + (k & 15) * 16 + o];
                else if (k < 272) v = nn_b2[(k - 256) * 16 + o];
                else              v = 0.0f;
                bfrag[kb][q] = (_Float16)v;
            }
        }
    }

    const int nTiles = Ee / 16;  // E divisible by 16
    const int wavesPerBlk = blockDim.x >> 5;
    const int waveId = blockIdx.x * wavesPerBlk + w;
    const int nWaves = gridDim.x * wavesPerBlk;

    for (int tile = waveId; tile < nTiles; tile += nWaves) {
        const int e = tile * 16 + m;
        // prefetch next tile's indices (global_prefetch_b8)
        if (tile + nWaves < nTiles) {
            __builtin_prefetch(&rowp[(tile + nWaves) * 16 + m], 0, 0);
            __builtin_prefetch(&colp[(tile + nWaves) * 16 + m], 0, 0);
        }
        _Float16 hhh[8];
        // ---- phase 1: both lane halves cooperate per edge: edge feats -> t (8 outputs each), gather h half
        {
            int r = rowp[e], c = colp[e];
            if (lane < 16) s_c[w][m] = c;
            int attr = edge_attr[e];
            float ef[12];
#pragma unroll
            for (int k = 0; k < 10; ++k) ef[k] = et_emb[attr * 10 + k];
            ef[10] = x[r * 3 + 0] - x[c * 3 + 0];
            ef[11] = x[r * 3 + 1] - x[c * 3 + 1];
#pragma unroll
            for (int q = 0; q < 8; ++q) hhh[q] = (_Float16)h0[r * 16 + half * 8 + q];
#pragma unroll
            for (int jj = 0; jj < 8; ++jj) {
                int j = half * 8 + jj;
                float a = s_b1[j];
#pragma unroll
                for (int k = 0; k < 12; ++k) a = fmaf(ef[k], s_w1[k * 16 + j], a);
                s_t[w][m][j] = eluf(a);
            }
        }
        __builtin_amdgcn_wave_barrier();   // keep LDS write->read order across phases

        // ---- phase 2: build A fragments with f16 packed math, 9 WMMA accumulations
        // A layout (16x32 f16): lane half 0 holds i=0..7 (k offsets 0..7 / 16..23),
        // half 1 holds i=8..15 (k offsets 8..15 / 24..31); j = 2kb (lo halfs), 2kb+1 (hi halfs).
        v8f acc = {};
#pragma unroll
        for (int kb = 0; kb < 8; ++kb) {
            _Float16 t0h = (_Float16)s_t[w][m][2 * kb];
            _Float16 t1h = (_Float16)s_t[w][m][2 * kb + 1];
            v16h afrag;
#pragma unroll
            for (int q = 0; q < 8; ++q) {
                afrag[q]     = t0h * hhh[q];
                afrag[q + 8] = t1h * hhh[q];
            }
            acc = __builtin_amdgcn_wmma_f32_16x16x32_f16(
                false, afrag, false, bfrag[kb], (short)0, acc, false, false);
        }
        {   // bias block (K = 256..287): U = h_i for k<272, zero pad
            v16h afrag;
#pragma unroll
            for (int q = 0; q < 8; ++q) {
                afrag[q]     = hhh[q];
                afrag[q + 8] = (_Float16)0.0f;
            }
            acc = __builtin_amdgcn_wmma_f32_16x16x32_f16(
                false, afrag, false, bfrag[8], (short)0, acc, false, false);
        }
        // ---- scatter: C layout: vgpr r -> M=r (lanes<16) / M=r+8 (lanes>=16), N=o
#pragma unroll
        for (int r8 = 0; r8 < 8; ++r8) {
            int M = r8 + half * 8;
            int cc = s_c[w][M];
            atomicAdd(&agg[cc * 16 + o], acc[r8]);
        }
    }
}

// ---------------------------------------------------------------- conv combine + LayerNorm + WMMA xw
// Per 16-node tile: every lane redundantly computes conv+LN+a1 for node (lane&15)
// -> A fragment (16x16 features, K padded to 32) held fully in registers, EXEC stays all-ones.
// B = gat_w staged block-wide in LDS as f16 [col][k] -> one aligned ds_load_b128 per fragment.
// 16x v_wmma_f32_16x16x32_f16 per tile (one per 16-column block of the 256 outputs).
__launch_bounds__(128)
__global__ void conv_xw_kernel(const float* __restrict__ h0, const float* __restrict__ agg,
                               const float* __restrict__ deg,
                               const float* __restrict__ conv_root, const float* __restrict__ conv_bias,
                               const float* __restrict__ a1, const float* __restrict__ gat_w,
                               float* __restrict__ xw) {
    __shared__ float    s_root[256];
    __shared__ float    s_cb[16];
    __shared__ _Float16 s_gw[256 * 16];   // [col][k] f16, 8KB

    const int tid = threadIdx.x;
    for (int i = tid; i < 256; i += blockDim.x) s_root[i] = conv_root[i];
    if (tid < 16) s_cb[tid] = conv_bias[tid];
    for (int idx = tid; idx < 4096; idx += blockDim.x) {
        int col = idx >> 4, k = idx & 15;
        s_gw[idx] = (_Float16)gat_w[k * 256 + col];
    }
    __syncthreads();

    const int lane = tid & 31;
    const int w    = tid >> 5;
    const int half = lane >> 4;
    const int m    = lane & 15;
    const int o    = lane & 15;
    const int koff = half * 8;

    const int nTiles = Nn / 16;   // N divisible by 16
    const int wavesPerBlk = blockDim.x >> 5;
    const int waveId = blockIdx.x * wavesPerBlk + w;
    const int nWaves = gridDim.x * wavesPerBlk;

    for (int tile = waveId; tile < nTiles; tile += nWaves) {
        const int n = tile * 16 + m;
        // ---- conv + ELU + LayerNorm (+a1) for node n, in registers
        float hin[16], hv[16];
#pragma unroll
        for (int j = 0; j < 16; ++j) hin[j] = h0[n * 16 + j];
        float dinv = 1.0f / fmaxf(deg[n], 1.0f);
#pragma unroll
        for (int oc = 0; oc < 16; ++oc) {
            float v = fmaf(agg[n * 16 + oc], dinv, s_cb[oc]);
#pragma unroll
            for (int j = 0; j < 16; ++j) v = fmaf(hin[j], s_root[j * 16 + oc], v);
            hv[oc] = eluf(v);
        }
        float mu = 0.0f;
#pragma unroll
        for (int oc = 0; oc < 16; ++oc) mu += hv[oc];
        mu *= (1.0f / 16.0f);
        float var = 0.0f;
#pragma unroll
        for (int oc = 0; oc < 16; ++oc) { float d = hv[oc] - mu; var = fmaf(d, d, var); }
        var *= (1.0f / 16.0f);
        float rs = rsqrtf(var + 1e-5f);
#pragma unroll
        for (int oc = 0; oc < 16; ++oc) hv[oc] = (hv[oc] - mu) * rs;
        if (n < Pp) {
#pragma unroll
            for (int oc = 0; oc < 16; ++oc) hv[oc] += a1[n * 16 + oc];
        }
        // ---- A fragment: lane half h holds K = koff..koff+7 (halfs 0..7), K+16 range is zero pad
        v16h afrag;
#pragma unroll
        for (int q = 0; q < 8; ++q) {
            afrag[q]     = (_Float16)hv[koff + q];
            afrag[q + 8] = (_Float16)0.0f;
        }
        // ---- 16 column blocks: B from LDS, one WMMA each, direct store
#pragma unroll 4
        for (int blk = 0; blk < 16; ++blk) {
            v16h bfrag;
            {                     // K=koff..koff+7 valid, K+16 range is zero pad
                int base = (blk * 16 + o) * 16 + koff;
#pragma unroll
                for (int q = 0; q < 8; ++q) {
                    bfrag[q]     = s_gw[base + q];
                    bfrag[q + 8] = (_Float16)0.0f;
                }
            }
            v8f c0 = {};
            v8f acc = __builtin_amdgcn_wmma_f32_16x16x32_f16(
                false, afrag, false, bfrag, (short)0, c0, false, false);
#pragma unroll
            for (int r8 = 0; r8 < 8; ++r8) {
                int M = r8 + half * 8;
                xw[(tile * 16 + M) * 256 + blk * 16 + o] = acc[r8];
            }
        }
    }
}

// ---------------------------------------------------------------- per-node attention coefficients
__global__ void attn_coef_kernel(const float* __restrict__ xw,
                                 const float* __restrict__ gat_as, const float* __restrict__ gat_ad,
                                 float* __restrict__ asrc, float* __restrict__ adst) {
    int i = blockIdx.x * blockDim.x + threadIdx.x;
    if (i >= Nn * 4) return;
    int n = i >> 2, hd = i & 3;
    float sa = 0.0f, da = 0.0f;
#pragma unroll 8
    for (int c = 0; c < 64; ++c) {
        float v = xw[n * 256 + hd * 64 + c];
        sa = fmaf(v, gat_as[hd * 64 + c], sa);
        da = fmaf(v, gat_ad[hd * 64 + c], da);
    }
    asrc[i] = sa;
    adst[i] = da;
}

// ---------------------------------------------------------------- attention pass A: segment max (only t<P matters)
__global__ void attn_max_kernel(const int* __restrict__ rowp, const int* __restrict__ colp,
                                const float* __restrict__ asrc, const float* __restrict__ adst,
                                unsigned* __restrict__ amaxk) {
    int i = blockIdx.x * blockDim.x + threadIdx.x;
    if (i >= Ee + Pp) return;
    int r, c;
    if (i < Ee) { r = rowp[i]; c = colp[i]; }
    else        { r = i - Ee; c = r; }          // self loops for n < P
    if (c >= Pp) return;
#pragma unroll
    for (int hd = 0; hd < 4; ++hd) {
        float alpha = leaky01(asrc[r * 4 + hd] + adst[c * 4 + hd]);
        atomicMax(&amaxk[c * 4 + hd], fkey(alpha));
    }
}

// ---------------------------------------------------------------- attention pass B: softmax accumulate (64 thr/edge)
__global__ void attn_acc_kernel(const int* __restrict__ rowp, const int* __restrict__ colp,
                                const float* __restrict__ asrc, const float* __restrict__ adst,
                                const unsigned* __restrict__ amaxk, const float* __restrict__ xw,
                                float* __restrict__ den, float* __restrict__ outacc) {
    int gid = blockIdx.x * blockDim.x + threadIdx.x;
    int ei = gid >> 6;
    int sub = gid & 63;
    if (ei >= Ee + Pp) return;
    int r, c;
    if (ei < Ee) { r = rowp[ei]; c = colp[ei]; }
    else         { r = ei - Ee; c = r; }
    if (c >= Pp) return;
    int col0 = sub * 4;
    int hd = col0 >> 6;
    float alpha = leaky01(asrc[r * 4 + hd] + adst[c * 4 + hd]);
    float ex = __expf(alpha - funkey(amaxk[c * 4 + hd]));
    if ((col0 & 63) == 0) atomicAdd(&den[c * 4 + hd], ex);
#pragma unroll
    for (int q = 0; q < 4; ++q) {
        int col = col0 + q;
        atomicAdd(&outacc[c * 256 + col], ex * xw[r * 256 + col]);
    }
}

// ---------------------------------------------------------------- final: head mean + bias + a2, net MLP -> y (P,1)
__launch_bounds__(64)
__global__ void final_kernel(const float* __restrict__ outacc, const float* __restrict__ den,
                             const float* __restrict__ gat_bias, const float* __restrict__ a2,
                             const float* __restrict__ net_w1, const float* __restrict__ net_b1,
                             const float* __restrict__ net_w2, const float* __restrict__ net_b2,
                             float* __restrict__ y) {
    int n = blockIdx.x * blockDim.x + threadIdx.x;
    if (n >= Pp) return;
    float dinv[4];
#pragma unroll
    for (int hd = 0; hd < 4; ++hd) dinv[hd] = 1.0f / den[n * 4 + hd];
    float z[64];
#pragma unroll
    for (int c = 0; c < 64; ++c) {
        float s = 0.0f;
#pragma unroll
        for (int hd = 0; hd < 4; ++hd) s = fmaf(outacc[n * 256 + hd * 64 + c], dinv[hd], s);
        z[c] = 0.25f * s + gat_bias[c] + a2[n * 64 + c];
    }
    float acc = net_b2[0];
    for (int j = 0; j < 64; ++j) {
        float s = net_b1[j];
#pragma unroll
        for (int c = 0; c < 64; ++c) s = fmaf(z[c], net_w1[c * 64 + j], s);
        acc = fmaf(eluf(s), net_w2[j], acc);
    }
    y[n] = acc;
}

// ================================================================ launcher
extern "C" void kernel_launch(void* const* d_in, const int* in_sizes, int n_in,
                              void* d_out, int out_size, void* d_ws, size_t ws_size,
                              hipStream_t stream) {
    const float* x        = (const float*)d_in[0];
    const float* action   = (const float*)d_in[1];
    const int*   ei       = (const int*)  d_in[2];
    const int*   eattr    = (const int*)  d_in[3];
    const int*   mask     = (const int*)  d_in[4];
    const float* pe_w1    = (const float*)d_in[5];
    const float* pe_b1    = (const float*)d_in[6];
    const float* pe_w2    = (const float*)d_in[7];
    const float* pe_b2    = (const float*)d_in[8];
    const float* ent_emb  = (const float*)d_in[9];
    const float* et_emb   = (const float*)d_in[10];
    const float* a1_w1    = (const float*)d_in[11];
    const float* a1_b1    = (const float*)d_in[12];
    const float* a1_w2    = (const float*)d_in[13];
    const float* a1_b2    = (const float*)d_in[14];
    const float* a2_w1    = (const float*)d_in[15];
    const float* a2_b1    = (const float*)d_in[16];
    const float* a2_w2    = (const float*)d_in[17];
    const float* a2_b2    = (const float*)d_in[18];
    const float* nn_w1    = (const float*)d_in[19];
    const float* nn_b1    = (const float*)d_in[20];
    const float* nn_w2    = (const float*)d_in[21];
    const float* nn_b2    = (const float*)d_in[22];
    const float* c1_root  = (const float*)d_in[23];
    const float* c1_bias  = (const float*)d_in[24];
    const float* gat_w    = (const float*)d_in[25];
    const float* gat_as   = (const float*)d_in[26];
    const float* gat_ad   = (const float*)d_in[27];
    const float* gat_bias = (const float*)d_in[28];
    const float* net_w1   = (const float*)d_in[29];
    const float* net_b1   = (const float*)d_in[30];
    const float* net_w2   = (const float*)d_in[31];
    const float* net_b2   = (const float*)d_in[32];

    const int* rowp = ei;
    const int* colp = ei + Ee;

    // workspace layout (floats); zeroed region first (atomics accumulate across graph replays)
    float*    ws     = (float*)d_ws;
    float*    w_deg  = ws;                               // N
    float*    w_agg  = w_deg + Nn;                       // N*16
    unsigned* w_amax = (unsigned*)(w_agg + Nn * 16);     // P*4 (ordered keys, 0 == -inf)
    float*    w_den  = (float*)(w_amax + Pp * 4);        // P*4
    float*    w_out  = w_den + Pp * 4;                   // P*256
    float*    w_a1   = w_out + Pp * 256;                 // P*16
    float*    w_a2   = w_a1 + Pp * 16;                   // P*64
    float*    w_h0   = w_a2 + Pp * 64;                   // N*16
    float*    w_xw   = w_h0 + Nn * 16;                   // N*256
    float*    w_asrc = w_xw + Nn * 256;                  // N*4
    float*    w_adst = w_asrc + Nn * 4;                  // N*4

    const int zeroN = Nn * 17 + Pp * 264;                // deg+agg+amax+den+outacc
    zero_kernel<<<(zeroN + 255) / 256, 256, 0, stream>>>(ws, zeroN);

    action_kernel<<<(Pp + 127) / 128, 128, 0, stream>>>(
        action, a1_w1, a1_b1, a1_w2, a1_b2, a2_w1, a2_b1, a2_w2, a2_b2, w_a1, w_a2);

    node_embed_kernel<<<(Nn + 255) / 256, 256, 0, stream>>>(
        x, mask, pe_w1, pe_b1, pe_w2, pe_b2, ent_emb, w_a1, w_h0);

    deg_kernel<<<(Ee + 255) / 256, 256, 0, stream>>>(colp, w_deg);

    edge_msg_kernel<<<1024, 128, 0, stream>>>(
        x, rowp, colp, eattr, et_emb, nn_w1, nn_b1, nn_w2, nn_b2, w_h0, w_agg);

    conv_xw_kernel<<<469, 128, 0, stream>>>(
        w_h0, w_agg, w_deg, c1_root, c1_bias, w_a1, gat_w, w_xw);

    attn_coef_kernel<<<(Nn * 4 + 255) / 256, 256, 0, stream>>>(
        w_xw, gat_as, gat_ad, w_asrc, w_adst);

    attn_max_kernel<<<(Ee + Pp + 255) / 256, 256, 0, stream>>>(
        rowp, colp, w_asrc, w_adst, w_amax);

    attn_acc_kernel<<<((Ee + Pp) * 64 + 255) / 256, 256, 0, stream>>>(
        rowp, colp, w_asrc, w_adst, w_amax, w_xw, w_den, w_out);

    final_kernel<<<(Pp + 63) / 64, 64, 0, stream>>>(
        w_out, w_den, gat_bias, w_a2, net_w1, net_b1, net_w2, net_b2, (float*)d_out);
}
